// MultiBondAttention_83743272337586
// MI455X (gfx1250) — matrix operand order repaired and checked by reading, not compiled, not run on hardware
//
#include <hip/hip_runtime.h>
#include <hip/hip_bf16.h>

typedef __attribute__((ext_vector_type(16))) _Float16 v16h;
typedef __attribute__((ext_vector_type(8)))  float    v8f;
typedef unsigned int u32x4 __attribute__((ext_vector_type(4)));
typedef int          i32x4 __attribute__((ext_vector_type(4)));
typedef int          i32x8 __attribute__((ext_vector_type(8)));

#define HID   256
#define NH    4
#define DH    64
#define LMAX  128
#define NMOLS 2048
#define LDX   264   // f16 row stride (256 + 8 pad) for X/Q/V/O tiles
#define LDK   136   // f16 row stride (128 + 8 pad) for K^T and P tiles

// Shared-memory layout (bytes). Total 307200 <= 320KB per WGP (CDNA5 ISA §3.1).
#define X_OFF   0                       // [128][LDX] f16  (X, later reused as O)
#define Q_OFF   67584                   // [128][LDX] f16
#define KT_OFF  135168                  // [NH][DH][LDK] f16 (K transposed per head)
#define V_OFF   204800                  // [128][LDX] f16
#define P_OFF   272384                  // [8 waves][16][LDK] f16 (attn probs scratch)
#define Z_OFF   67584                   // [128][256] f32 overlay of Q+KT (dead regions)
#define STG_OFF Z_OFF                   // f32 staging for the TDM gather (same overlay)
#define SMEM_BYTES 307200

// Tensor Data Mover path: one DMA pulls the whole ragged [size x 256] f32 tile
// into LDS; rows >= tensor_dim1 come back as zeros (ISA §8.2 OOB rule), which
// implements the padding mask for free.
#if defined(__AMDGCN__) && __has_builtin(__builtin_amdgcn_tensor_load_to_lds) && \
    __has_builtin(__builtin_amdgcn_s_wait_tensorcnt)
#define USE_TDM 1
#else
#define USE_TDM 0
#endif

static __device__ __forceinline__ v8f vzero8() {
  v8f z = {0.f, 0.f, 0.f, 0.f, 0.f, 0.f, 0.f, 0.f};
  return z;
}

static __device__ __forceinline__ v8f wmma16(v16h a, v16h b, v8f c) {
  // D = A(16x32 f16) * B(32x16 f16) + C(16x16 f32)
  return __builtin_amdgcn_wmma_f32_16x16x32_f16(
      /*neg_a=*/false, a, /*neg_b=*/false, b,
      /*c_mod=*/(short)0, c, /*reuse_a=*/false, /*reuse_b=*/false);
}

// A fragment (16x32 f16) from LDS row-major [row][k] with leading dim ld.
// ISA 7.12.2: lane L<16 holds row L; K pairs are contiguous -> ds_load_b128.
static __device__ __forceinline__ v16h load_a_frag(const _Float16* base, int ld,
                                                   int row0, int k0) {
  int lane = threadIdx.x & 31;
  int r = lane & 15, hf = lane >> 4;
  const _Float16* rp = base + (row0 + r) * ld + k0 + 16 * hf;
  v16h a;
#pragma unroll
  for (int p = 0; p < 8; ++p) {
    int k = 2 * p;
    a[2 * p]     = rp[k];
    a[2 * p + 1] = rp[k + 1];
  }
  return a;
}

// B fragment (32x16 f16): element (k,n) = base[(k0+k)*ld + n0+n].
static __device__ __forceinline__ v16h load_b_frag_lds(const _Float16* base, int ld,
                                                       int k0, int n0) {
  int lane = threadIdx.x & 31;
  int n = lane & 15, hf = lane >> 4;
  const _Float16* bp = base + (k0 + 16 * hf) * ld + n0 + n;
  v16h b;
#pragma unroll
  for (int p = 0; p < 8; ++p) {
    b[2 * p]     = bp[(2 * p) * ld];
    b[2 * p + 1] = bp[(2 * p + 1) * ld];
  }
  return b;
}

// B fragment for Y = X @ W^T: B(k,n) = W[n0+n][k0+k], W row-major f32 [HID][HID].
static __device__ __forceinline__ v16h load_b_frag_w(const float* __restrict__ W,
                                                     int n0, int k0) {
  int lane = threadIdx.x & 31;
  int n = lane & 15, hf = lane >> 4;
  const float* wp = W + (size_t)(n0 + n) * HID + k0 + 16 * hf;
  v16h b;
#pragma unroll
  for (int i = 0; i < 16; i += 4) {
    float4 v = *reinterpret_cast<const float4*>(wp + i);
    b[i]     = (_Float16)v.x;
    b[i + 1] = (_Float16)v.y;
    b[i + 2] = (_Float16)v.z;
    b[i + 3] = (_Float16)v.w;
  }
  return b;
}

// C tile (16x16 f32) -> LDS f16.
static __device__ __forceinline__ void store_c_f16(v8f c, _Float16* base, int ld,
                                                   int row0, int col0, bool relu) {
  int lane = threadIdx.x & 31;
  int n = lane & 15, hf = lane >> 4;
  _Float16* p = base + (row0 + 8 * hf) * ld + col0 + n;
#pragma unroll
  for (int r = 0; r < 8; ++r) {
    float v = c[r];
    if (relu) v = fmaxf(v, 0.f);
    p[r * ld] = (_Float16)v;
  }
}

__global__ void __launch_bounds__(256)
mba_kernel(const float* __restrict__ message,
           const float* __restrict__ Wq, const float* __restrict__ Wk,
           const float* __restrict__ Wv, const float* __restrict__ Wo,
           const float* __restrict__ bo, const float* __restrict__ gamma,
           const float* __restrict__ beta,
           const int* __restrict__ b_starts, const int* __restrict__ b_sizes,
           float* __restrict__ out) {
  extern __shared__ __align__(16) char smem[];
  _Float16* sX  = reinterpret_cast<_Float16*>(smem + X_OFF);
  _Float16* sQ  = reinterpret_cast<_Float16*>(smem + Q_OFF);
  _Float16* sKT = reinterpret_cast<_Float16*>(smem + KT_OFF);
  _Float16* sV  = reinterpret_cast<_Float16*>(smem + V_OFF);
  _Float16* sP  = reinterpret_cast<_Float16*>(smem + P_OFF);
  float*    sZ  = reinterpret_cast<float*>(smem + Z_OFF);

  const int tid  = threadIdx.x;
  const int lane = tid & 31;
  const int wave = tid >> 5;
  const int start = b_starts[blockIdx.x];
  const int size  = b_sizes[blockIdx.x];

  // ---------- Stage 1: ragged gather into LDS as f16 ----------
#if USE_TDM
  {
    if (wave == 0) {
      // D# group 0 (ISA §8.3): count=1, lds_addr, 57-bit global_addr, type=2.
      unsigned long long ga =
          (unsigned long long)(uintptr_t)message +
          (unsigned long long)start * (unsigned long long)(HID * 4);
      // Generic LDS pointer: addr[31:0] is the LDS byte offset (ISA §10.2).
      unsigned lds_base = (unsigned)(uintptr_t)(smem + STG_OFF);
      u32x4 g0;
      g0[0] = 1u;                                     // count=1, no gather
      g0[1] = lds_base;                               // lds_addr
      g0[2] = (unsigned)ga;                           // global_addr[31:0]
      g0[3] = (unsigned)((ga >> 32) & 0x01FFFFFFull)  // global_addr[56:32]
              | (2u << 30);                           // type=2 ("image")
      // D# group 1 (ISA §8.4): data_size=4B, tensor 256 x size, tile 256 x 128,
      // dim0 stride 256. Rows >= size read back zero (OOB), no padding ops.
      i32x8 g1;
      g1[0] = (int)(2u << 16);            // data_size=2 (4 bytes), wg_mask=0
      g1[1] = (int)(256u << 16);          // tensor_dim0[15:0] @ bits 63:48
      g1[2] = (int)((unsigned)size << 16);// tensor_dim1[15:0] @ bits 95:80
      g1[3] = (int)(256u << 16);          // tile_dim0=256     @ bits 127:112
      g1[4] = 128;                        // tile_dim1=128     @ bits 143:128
      g1[5] = 256;                        // tensor_dim0_stride[31:0]
      g1[6] = 0;
      g1[7] = 0;
      i32x4 z4; z4[0] = 0; z4[1] = 0; z4[2] = 0; z4[3] = 0;
#if __clang_major__ >= 23
      i32x8 z8; 
#pragma unroll
      for (int i = 0; i < 8; ++i) z8[i] = 0;
      __builtin_amdgcn_tensor_load_to_lds(g0, g1, z4, z4, z8, 0);
#else
      __builtin_amdgcn_tensor_load_to_lds(g0, g1, z4, z4, 0);
#endif
    }
    __builtin_amdgcn_s_wait_tensorcnt(0);
    __syncthreads();
    // Convert staged f32 tile -> f16 sX (padded rows already zero via TDM OOB).
    int row  = tid >> 1;
    int half = tid & 1;
    const float* src = reinterpret_cast<const float*>(smem + STG_OFF) +
                       row * HID + half * 128;
    _Float16* dst = sX + row * LDX + half * 128;
#pragma unroll
    for (int j = 0; j < 128; j += 4) {
      float4 v = *reinterpret_cast<const float4*>(src + j);
      dst[j]     = (_Float16)v.x;
      dst[j + 1] = (_Float16)v.y;
      dst[j + 2] = (_Float16)v.z;
      dst[j + 3] = (_Float16)v.w;
    }
  }
#else
  {
    int row  = tid >> 1;
    int half = tid & 1;
    _Float16* dst = sX + row * LDX + half * 128;
    if (row < size) {
      const float* src = message + (size_t)(start + row) * HID + half * 128;
#pragma unroll
      for (int j = 0; j < 128; j += 4) {
        float4 v = *reinterpret_cast<const float4*>(src + j);
        dst[j]     = (_Float16)v.x;
        dst[j + 1] = (_Float16)v.y;
        dst[j + 2] = (_Float16)v.z;
        dst[j + 3] = (_Float16)v.w;
      }
    } else {
#pragma unroll
      for (int j = 0; j < 128; ++j) dst[j] = (_Float16)0.f;
    }
  }
#endif
  __syncthreads();

  // ---------- Stage 2: Q/K/V = X @ W^T via WMMA ----------
  // Wave owns 2 column tiles across all 8 row tiles: one W fragment feeds 8 WMMAs.
#pragma unroll 1
  for (int mat = 0; mat < 3; ++mat) {
    const float* W = (mat == 0) ? Wq : ((mat == 1) ? Wk : Wv);
#pragma unroll 1
    for (int ntl = 0; ntl < 2; ++ntl) {
      int nt = wave * 2 + ntl;
      v8f acc[8];
#pragma unroll
      for (int mt = 0; mt < 8; ++mt) acc[mt] = vzero8();
#pragma unroll 1
      for (int kk = 0; kk < 8; ++kk) {
        // Speculative prefetch of the next K-slice (dropped silently if OOB).
        __builtin_prefetch(W + (size_t)(nt * 16 + (lane & 15)) * HID + kk * 32 + 32, 0, 1);
        v16h b = load_b_frag_w(W, nt * 16, kk * 32);
#pragma unroll
        for (int mt = 0; mt < 8; ++mt) {
          v16h a = load_a_frag(sX, LDX, mt * 16, kk * 32);
          acc[mt] = wmma16(a, b, acc[mt]);
        }
      }
      if (mat == 1) {
        // K tile -> transposed per-head store: sKT[h][d][k_bond]
        int h = nt >> 2;
        int dbase = (nt & 3) * 16;
        _Float16* kt = sKT + h * (DH * LDK);
        int n = lane & 15, hf = lane >> 4;
#pragma unroll
        for (int mt = 0; mt < 8; ++mt) {
          _Float16* p = kt + (dbase + n) * LDK + mt * 16 + 8 * hf;
#pragma unroll
          for (int r = 0; r < 8; ++r) p[r] = (_Float16)acc[mt][r];
        }
      } else {
        _Float16* dstm = (mat == 0) ? sQ : sV;
#pragma unroll
        for (int mt = 0; mt < 8; ++mt)
          store_c_f16(acc[mt], dstm, LDX, mt * 16, nt * 16, false);
      }
    }
  }
  __syncthreads();

  // ---------- Stage 3: attention. Wave w owns query tile w, loops heads ----------
  const int qr0 = wave * 16;
  _Float16* sPw = sP + wave * (16 * LDK);
  _Float16* sO  = sX;               // X region is dead; reuse for O
  const float scale = 0.125f;       // DH^-0.5
#pragma unroll 1
  for (int h = 0; h < NH; ++h) {
    const _Float16* kth = sKT + h * (DH * LDK);
    v8f s[8];
#pragma unroll
    for (int nt = 0; nt < 8; ++nt) s[nt] = vzero8();
#pragma unroll 1
    for (int kk = 0; kk < 2; ++kk) {           // K-dim = DH = 64 -> 2 steps of 32
      v16h a = load_a_frag(sQ, LDX, qr0, h * DH + kk * 32);
#pragma unroll
      for (int nt = 0; nt < 8; ++nt) {         // 8 key tiles of 16
        v16h b = load_b_frag_lds(kth, LDK, kk * 32, nt * 16);
        s[nt] = wmma16(a, b, s[nt]);
      }
    }
    // Masked softmax across the 128 key columns for 16 query rows.
    {
      int nlow = lane & 15;
      int hf = lane >> 4;
#pragma unroll
      for (int r = 0; r < 8; ++r) {
        float vals[8];
        float mx = -3.0e38f;
#pragma unroll
        for (int nt = 0; nt < 8; ++nt) {
          int kcol = nt * 16 + nlow;
          float v = (kcol < size) ? s[nt][r] * scale : -3.0e38f;
          vals[nt] = v;
          mx = fmaxf(mx, v);
        }
#pragma unroll
        for (int d = 1; d < 16; d <<= 1) mx = fmaxf(mx, __shfl_xor(mx, d, 32));
        float sum = 0.f;
#pragma unroll
        for (int nt = 0; nt < 8; ++nt) {
          int kcol = nt * 16 + nlow;
          float p = (kcol < size) ? __expf(vals[nt] - mx) : 0.f;
          vals[nt] = p;
          sum += p;
        }
#pragma unroll
        for (int d = 1; d < 16; d <<= 1) sum += __shfl_xor(sum, d, 32);
        float inv = 1.f / sum;
        int row = r + 8 * hf;
#pragma unroll
        for (int nt = 0; nt < 8; ++nt)
          sPw[row * LDK + nt * 16 + nlow] = (_Float16)(vals[nt] * inv);
      }
    }
    // O_h = P @ V_h, ReLU, write into sO (this wave's own 16 rows).
    {
      v8f o[4];
#pragma unroll
      for (int nt = 0; nt < 4; ++nt) o[nt] = vzero8();
#pragma unroll 1
      for (int kk = 0; kk < 4; ++kk) {         // K-dim = 128 -> 4 steps of 32
        v16h a = load_a_frag(sPw, LDK, 0, kk * 32);
#pragma unroll
        for (int nt = 0; nt < 4; ++nt) {       // DH = 64 -> 4 output tiles
          v16h b = load_b_frag_lds(sV, LDX, kk * 32, h * DH + nt * 16);
          o[nt] = wmma16(a, b, o[nt]);
        }
      }
#pragma unroll
      for (int nt = 0; nt < 4; ++nt)
        store_c_f16(o[nt], sO, LDX, qr0, h * DH + nt * 16, true);
    }
  }
  __syncthreads();

  // ---------- Stage 4: Z = O @ Wo^T + bo into f32 LDS overlay ----------
#pragma unroll 1
  for (int ntl = 0; ntl < 2; ++ntl) {
    int nt = wave * 2 + ntl;
    v8f acc[8];
#pragma unroll
    for (int mt = 0; mt < 8; ++mt) acc[mt] = vzero8();
#pragma unroll 1
    for (int kk = 0; kk < 8; ++kk) {
      v16h b = load_b_frag_w(Wo, nt * 16, kk * 32);
#pragma unroll
      for (int mt = 0; mt < 8; ++mt) {
        v16h a = load_a_frag(sO, LDX, mt * 16, kk * 32);
        acc[mt] = wmma16(a, b, acc[mt]);
      }
    }
    int n = lane & 15, hf = lane >> 4;
    float bias = bo[nt * 16 + n];
#pragma unroll
    for (int mt = 0; mt < 8; ++mt) {
      float* zp = sZ + (mt * 16 + 8 * hf) * HID + nt * 16 + n;
#pragma unroll
      for (int r = 0; r < 8; ++r) zp[r * HID] = acc[mt][r] + bias;
    }
  }
  __syncthreads();

  // ---------- Stage 5: LayerNorm + masked scatter ----------
  {
    int row = wave * 16 + (lane >> 1);       // 2 lanes per row
    int c0  = (lane & 1) * 128;
    const float* zr = sZ + row * HID;
    float s1 = 0.f, s2 = 0.f;
#pragma unroll
    for (int j = 0; j < 128; j += 4) {
      float4 v = *reinterpret_cast<const float4*>(zr + c0 + j);
      s1 += v.x + v.y + v.z + v.w;
      s2 += v.x * v.x + v.y * v.y + v.z * v.z + v.w * v.w;
    }
    s1 += __shfl_xor(s1, 1, 32);
    s2 += __shfl_xor(s2, 1, 32);
    float mu   = s1 * (1.f / 256.f);
    float var  = s2 * (1.f / 256.f) - mu * mu;
    float rstd = rsqrtf(var + 1e-5f);
    if (row < size) {
      float* orow = out + (size_t)(start + row) * HID + c0;
#pragma unroll
      for (int j = 0; j < 128; j += 4) {
        float4 v  = *reinterpret_cast<const float4*>(zr + c0 + j);
        float4 g  = *reinterpret_cast<const float4*>(gamma + c0 + j);
        float4 bb = *reinterpret_cast<const float4*>(beta + c0 + j);
        float4 o;
        o.x = (v.x - mu) * rstd * g.x + bb.x;
        o.y = (v.y - mu) * rstd * g.y + bb.y;
        o.z = (v.z - mu) * rstd * g.z + bb.z;
        o.w = (v.w - mu) * rstd * g.w + bb.w;
        *reinterpret_cast<float4*>(orow + j) = o;
      }
    }
  }
}

// d_out is poisoned by the harness; row 0 and any unwritten elements must be 0.
__global__ void zero_f32(float* __restrict__ p, long long n) {
  long long i = (long long)blockIdx.x * blockDim.x + threadIdx.x;
  long long stride = (long long)gridDim.x * blockDim.x;
  for (; i < n; i += stride) p[i] = 0.f;
}

extern "C" void kernel_launch(void* const* d_in, const int* in_sizes, int n_in,
                              void* d_out, int out_size, void* d_ws, size_t ws_size,
                              hipStream_t stream) {
  (void)n_in; (void)d_ws; (void)ws_size; (void)in_sizes;
  const float* message = (const float*)d_in[0];
  const float* Wq      = (const float*)d_in[1];
  const float* Wk      = (const float*)d_in[2];
  const float* Wv      = (const float*)d_in[3];
  const float* Wo      = (const float*)d_in[4];
  const float* bo      = (const float*)d_in[5];
  const float* gamma   = (const float*)d_in[6];
  const float* beta    = (const float*)d_in[7];
  const int*   b_starts = (const int*)d_in[8];
  const int*   b_sizes  = (const int*)d_in[9];
  float* out = (float*)d_out;

  // Allow >default dynamic LDS (CDNA5 WGP has 320KB). Host-side attribute set;
  // not a stream operation, deterministic, graph-capture safe.
  (void)hipFuncSetAttribute((const void*)mba_kernel,
                            hipFuncAttributeMaxDynamicSharedMemorySize,
                            SMEM_BYTES);

  zero_f32<<<2048, 256, 0, stream>>>(out, (long long)out_size);
  mba_kernel<<<NMOLS, 256, SMEM_BYTES, stream>>>(message, Wq, Wk, Wv, Wo, bo,
                                                 gamma, beta, b_starts, b_sizes,
                                                 out);
}
